// DGCNN_18975165514164
// MI455X (gfx1250) — matrix operand order repaired and checked by reading, not compiled, not run on hardware
//
#include <hip/hip_runtime.h>

// ---------------------------------------------------------------------------
// DGCNN forward for MI455X (gfx1250, wave32).
//  - All 1x1-conv GEMMs: v_wmma_f32_16x16x32_bf16 (bf16 in, f32 acc); fc layers scalar fp32
//  - Pairwise-distance Gram matrices: v_wmma_f32_16x16x4_f32 (full fp32)
//  - Dense-B GEMMs stage their [32K x 64P] B tile into LDS with
//    global_load_async_to_lds_b128 (+ s_wait_asynccnt), then feed WMMA from LDS.
//  - max-over-k / max-over-N folded into GEMM epilogue via monotonicity of
//    BN-normalize + leaky-relu, using order-preserving uint-encoded atomicMax.
// ---------------------------------------------------------------------------

#define B_   8
#define N_   1024
#define KK_  20
#define EPSV 1e-5f

typedef __attribute__((ext_vector_type(2)))  float   v2f;
typedef __attribute__((ext_vector_type(8)))  float   v8f;
typedef __attribute__((ext_vector_type(16))) __bf16  v16bf;

static __device__ __forceinline__ unsigned short f2bf(float f) {
    unsigned u = __float_as_uint(f);
    unsigned r = u + 0x7FFFu + ((u >> 16) & 1u);   // round-to-nearest-even
    return (unsigned short)(r >> 16);
}
static __device__ __forceinline__ __bf16 bfbits(unsigned short s) {
    union { unsigned short u; __bf16 b; } x; x.u = s; return x.b;
}
// order-preserving float <-> uint encoding (monotone for all finite floats)
static __device__ __forceinline__ unsigned encf(float f) {
    unsigned u = __float_as_uint(f);
    return (u & 0x80000000u) ? ~u : (u | 0x80000000u);
}
static __device__ __forceinline__ float decf(unsigned e) {
    return __uint_as_float((e & 0x80000000u) ? (e & 0x7FFFFFFFu) : ~e);
}
static __device__ __forceinline__ float lrelu(float v) { return v > 0.f ? v : 0.2f * v; }

// wave-relative 32-bit LDS byte offset of a __shared__ object (generic -> AS3)
static __device__ __forceinline__ unsigned lds_addr32(void* p) {
    return (unsigned)(unsigned long long)(__attribute__((address_space(3))) char*)p;
}
// async 16-byte global -> LDS copy (no VGPR data; tracked by ASYNCcnt)
static __device__ __forceinline__ void async_copy_b128(unsigned lds_off, const void* gaddr) {
    asm volatile("global_load_async_to_lds_b128 %0, %1, off"
                 :: "v"(lds_off), "v"((unsigned long long)gaddr) : "memory");
}
static __device__ __forceinline__ void wait_async0() {
    asm volatile("s_wait_asynccnt 0x0" ::: "memory");
}

// --------------------------- small utility kernels -------------------------

__global__ void cvt_bf16(const float* __restrict__ in, unsigned short* __restrict__ out, int n) {
    int i = blockIdx.x * blockDim.x + threadIdx.x;
    if (i < n) out[i] = f2bf(in[i]);
}

// per-point squared norm: xx[b,n] = sum_c x[b,c0+c,n]^2   (x strided by Cs)
__global__ void xx_kernel(const float* __restrict__ xb, int c0, int Cs, int C,
                          float* __restrict__ xx) {
    int i = blockIdx.x * blockDim.x + threadIdx.x;
    if (i >= B_ * N_) return;
    int b = i / N_, n = i - b * N_;
    const float* xc = xb + ((size_t)b * Cs + c0) * N_ + n;
    float s = 0.f;
    for (int c = 0; c < C; ++c) { float v = xc[(size_t)c * N_]; s += v * v; }
    xx[i] = s;
}

// ------------------- fp32 WMMA Gram / pairwise distance --------------------
// pd[b,n,m] = 2 * <x_n, x_m> - ||x_n||^2 - ||x_m||^2
template<bool CFULL>
__global__ void gram_wmma(const float* __restrict__ xb, int c0, int Cs, int C,
                          const float* __restrict__ xx, float* __restrict__ pd) {
    const int lane = threadIdx.x & 31;
    const int half = lane >> 4, lm = lane & 15;
    const int n0 = blockIdx.x * 16, m0 = blockIdx.y * 16, b = blockIdx.z;
    const float* xpb = xb + ((size_t)b * Cs + c0) * N_;
    const float* pa  = xpb + (n0 + lm);     // A column (row n0+lm of x^T)
    const float* pb  = xpb + (m0 + lm);     // B column
    v8f acc = {};
    for (int c = 0; c < C; c += 4) {
        v2f a, bb;
#pragma unroll
        for (int e = 0; e < 2; ++e) {
            int k = c + e + 2 * half;       // f32 A/B 16x4 layout
            if (CFULL) {
                a[e]  = pa[(size_t)k * N_];
                bb[e] = pb[(size_t)k * N_];
            } else {
                a[e]  = (k < C) ? pa[(size_t)k * N_] : 0.f;
                bb[e] = (k < C) ? pb[(size_t)k * N_] : 0.f;
            }
        }
        acc = __builtin_amdgcn_wmma_f32_16x16x4_f32(false, a, false, bb,
                                                    (short)0, acc, false, false);
    }
    const float xxm = xx[b * N_ + m0 + lm];
    float* prow = pd + ((size_t)b * N_ + n0) * N_ + m0 + lm;
#pragma unroll
    for (int r = 0; r < 8; ++r) {
        int n = r + 8 * half;               // D: M = r + 8*(lane/16), N = lane%16
        prow[(size_t)n * N_] = 2.f * acc[r] - xx[b * N_ + n0 + n] - xxm;
    }
}

// ------------------------------- top-k (k=20) ------------------------------
__global__ void topk20(const float* __restrict__ pd, int* __restrict__ idx) {
    __shared__ float sv[N_];
    __shared__ float rb[128];
    __shared__ int   ri[128];
    const int row = blockIdx.x;                    // b*N + n
    const int tid = threadIdx.x;
    const float* prow = pd + (size_t)row * N_;
    for (int i = tid; i < N_; i += 128) sv[i] = prow[i];
    __syncthreads();
    for (int t = 0; t < KK_; ++t) {
        float bv = -__builtin_inff(); int bi = N_;
        for (int i = tid; i < N_; i += 128) {
            float v = sv[i];
            if (v > bv || (v == bv && i < bi)) { bv = v; bi = i; }
        }
        rb[tid] = bv; ri[tid] = bi;
        __syncthreads();
        for (int s = 64; s > 0; s >>= 1) {
            if (tid < s) {
                float ov = rb[tid + s]; int oi = ri[tid + s];
                if (ov > rb[tid] || (ov == rb[tid] && oi < ri[tid])) { rb[tid] = ov; ri[tid] = oi; }
            }
            __syncthreads();
        }
        if (tid == 0) { idx[row * KK_ + t] = ri[0]; sv[ri[0]] = -__builtin_inff(); }
        __syncthreads();
    }
}

// ------------------ shared GEMM epilogue (stats + max/store) ---------------
template<int OUTM>
static __device__ __forceinline__ void gemm_epilogue(
        const v8f& acc, int O, int o0, int half, int lm,
        int eb, int en, int ej,
        unsigned* __restrict__ emax, float* __restrict__ yfull,
        float* __restrict__ s_sum, float* __restrict__ s_sq) {
#pragma unroll
    for (int r = 0; r < 8; ++r) {
        int o = o0 + r + 8 * half;
        float y = acc[r];
        float s = y, q = y * y;
#pragma unroll
        for (int m = 1; m < 16; m <<= 1) {
            s += __shfl_xor(s, m, 32);
            q += __shfl_xor(q, m, 32);
        }
        if (lm == 0) {
            atomicAdd(&s_sum[o], s);
            atomicAdd(&s_sq[o], q);
        }
        if (OUTM == 0) {
            atomicMax(&emax[((size_t)eb * O + o) * N_ + en], encf(y));
        } else if (OUTM == 1) {
            atomicMax(&emax[(size_t)eb * O + o], encf(y));
        } else {
            yfull[(((size_t)eb * O + o) * N_ + en) * KK_ + ej] = y;
        }
    }
}

// --------------- bf16 WMMA GEMM, gathered edge features (B) ----------------
// Y[O,P] = W[O,Kc] @ G[Kc,P], G[k,p] = edge feature, P = B*N*KK
// KFULL: Kc % 32 == 0.  OUTM as in gemm_epilogue.
template<int OUTM, bool KFULL>
__global__ void edge_gemm(const unsigned short* __restrict__ wbf, int O, int Kc,
                          const float* __restrict__ xbase, int c0, int Cs,
                          const int* __restrict__ nbr,
                          unsigned* __restrict__ emax,
                          float* __restrict__ yfull,
                          float* __restrict__ s_sum, float* __restrict__ s_sq) {
    const int lane = threadIdx.x & 31;
    const int wv   = threadIdx.x >> 5;
    const int half = lane >> 4, lm = lane & 15;
    const int p0 = blockIdx.x * 64 + wv * 16;      // 4 waves split the P tile
    const int o0 = blockIdx.y * 16;
    const int C  = Kc >> 1;

    // hoist per-column decode out of the K loop (loop-invariant)
    int ofsC[16], ofsN[16];
#pragma unroll
    for (int i = 0; i < 16; ++i) {
        int p = p0 + i;
        int b  = p / (N_ * KK_);
        int r1 = p - b * N_ * KK_;
        int n  = r1 / KK_, j = r1 - n * KK_;
        int xcb = (b * Cs + c0) * N_;
        ofsC[i] = xcb + n;
        ofsN[i] = xcb + nbr[(b * N_ + n) * KK_ + j];
    }
    const unsigned short* wrow = wbf + (size_t)(o0 + lm) * Kc;

    v8f acc = {};
    for (int kk = 0; kk < Kc; kk += 32) {
        v16bf af;
        if (KFULL) {
            union { uint4 u; unsigned short s[8]; } wlo, whi;
            wlo.u = *(const uint4*)(wrow + kk + 8 * half);
            whi.u = *(const uint4*)(wrow + kk + 16 + 8 * half);
#pragma unroll
            for (int i = 0; i < 8; ++i) {
                af[i]     = bfbits(wlo.s[i]);
                af[8 + i] = bfbits(whi.s[i]);
            }
            if (kk + 32 < Kc) __builtin_prefetch(wrow + kk + 32, 0, 1);
        } else {
#pragma unroll
            for (int i = 0; i < 16; ++i) {
                int k = kk + i + 8 * half + (i >= 8 ? 8 : 0);
                af[i] = (k < Kc) ? bfbits(wrow[k]) : bfbits(0);
            }
        }
        v16bf bf;
        {
            const int oN = ofsN[lm], oC = ofsC[lm];
#pragma unroll
            for (int i = 0; i < 16; ++i) {
                const int k = kk + i + 8 * half + (i >= 8 ? 8 : 0);
                if (k < C)
                    bf[i] = bfbits(f2bf(xbase[oN + k * N_] - xbase[oC + k * N_]));
                else if (k < Kc)
                    bf[i] = bfbits(f2bf(xbase[oC + (k - C) * N_]));
                else
                    bf[i] = bfbits(0);
            }
        }
        acc = __builtin_amdgcn_wmma_f32_16x16x32_bf16(false, af, false, bf,
                                                      (short)0, acc, false, false);
    }

    int p = p0 + lm;
    int eb = p / (N_ * KK_);
    int r1 = p - eb * N_ * KK_;
    int en = r1 / KK_, ej = r1 - en * KK_;
    gemm_epilogue<OUTM>(acc, O, o0, half, lm, eb, en, ej, emax, yfull, s_sum, s_sq);
}

// -------- bf16 WMMA GEMM, dense B staged through LDS via async copies ------
// INM=1: B rows from 4D tensor [B,Kc,N,KK] (row stride N*KK), P = B*N*KK
// INM=2: B rows from 3D tensor (src,c0,Cs) (row stride N),    P = B*N
// Kc must be a multiple of 32.  Per K step the block DMAs a [32 x 64] f32
// tile into LDS with global_load_async_to_lds_b128, waits on ASYNCcnt,
// and feeds the WMMA B fragments from LDS (68-float row pitch: conflict-free).
template<int INM, int OUTM>
__global__ void dense_gemm(const unsigned short* __restrict__ wbf, int O, int Kc,
                           const float* __restrict__ src, int c0, int Cs,
                           unsigned* __restrict__ emax,
                           float* __restrict__ s_sum, float* __restrict__ s_sq) {
    __shared__ float tile[32 * 68];                // 8704 B, padded pitch
    const int lane = threadIdx.x & 31;
    const int wv   = threadIdx.x >> 5;
    const int half = lane >> 4, lm = lane & 15;
    const int p0b  = blockIdx.x * 64;              // block-level P tile
    const int p0   = p0b + wv * 16;
    const int o0   = blockIdx.y * 16;

    const int rowStride = (INM == 2) ? N_ : (N_ * KK_);
    const int pb = p0b / ((INM == 2) ? N_ : (N_ * KK_));   // tile-uniform batch
    size_t cbase;
    if (INM == 2) cbase = ((size_t)pb * Cs + c0) * N_ + (p0b - pb * N_);
    else          cbase = (size_t)pb * Kc * (N_ * KK_) + (size_t)(p0b - pb * (N_ * KK_));

    const unsigned short* wrow = wbf + (size_t)(o0 + lm) * Kc;
    // this thread's 4 staging chunks: chunk id -> (row, 4-float column)
    const int tid = threadIdx.x;

    v8f acc = {};
    for (int kk = 0; kk < Kc; kk += 32) {
        // ---- async-stage B tile [32 rows x 64 floats] into LDS ----
#pragma unroll
        for (int t = 0; t < 4; ++t) {
            int id  = tid + t * 128;               // 0..511
            int row = id >> 4;                     // 0..31
            int cf  = (id & 15) << 2;              // 0,4,...,60 floats
            const float* gp = src + cbase + (size_t)(kk + row) * rowStride + cf;
            async_copy_b128(lds_addr32(&tile[row * 68 + cf]), gp);
        }
        wait_async0();
        __syncthreads();

        // ---- A fragment: weights (two 16B vector loads) ----
        v16bf af;
        {
            union { uint4 u; unsigned short s[8]; } wlo, whi;
            wlo.u = *(const uint4*)(wrow + kk + 8 * half);
            whi.u = *(const uint4*)(wrow + kk + 16 + 8 * half);
#pragma unroll
            for (int i = 0; i < 8; ++i) {
                af[i]     = bfbits(wlo.s[i]);
                af[8 + i] = bfbits(whi.s[i]);
            }
            if (kk + 32 < Kc) __builtin_prefetch(wrow + kk + 32, 0, 1);
        }
        // ---- B fragment from LDS: lane%16 -> column, element -> K row (table order) ----
        v16bf bf;
        {
            const float* lr = &tile[wv * 16 + lm];
#pragma unroll
            for (int i = 0; i < 16; ++i)
                bf[i] = bfbits(f2bf(lr[(i + 8 * half + (i >= 8 ? 8 : 0)) * 68]));
        }
        acc = __builtin_amdgcn_wmma_f32_16x16x32_bf16(false, af, false, bf,
                                                      (short)0, acc, false, false);
        __syncthreads();                           // before next tile overwrite
    }

    int p = p0 + lm;
    int eb, en, ej;
    if (INM == 2) { eb = p / N_; en = p - eb * N_; ej = 0; }
    else {
        eb = p / (N_ * KK_);
        int r1 = p - eb * N_ * KK_;
        en = r1 / KK_; ej = r1 - en * KK_;
    }
    gemm_epilogue<OUTM>(acc, O, o0, half, lm, eb, en, ej, emax, nullptr, s_sum, s_sq);
}

// ---------------------------- BN finalize / apply --------------------------

__global__ void finalize_stats(const float* __restrict__ s_sum, const float* __restrict__ s_sq,
                               float cnt, float* __restrict__ meanv, float* __restrict__ invstd,
                               int O) {
    int o = blockIdx.x * blockDim.x + threadIdx.x;
    if (o >= O) return;
    float m = s_sum[o] / cnt;
    float v = s_sq[o] / cnt - m * m;
    meanv[o]  = m;
    invstd[o] = rsqrtf(fmaxf(v, 0.f) + EPSV);
}

__global__ void norm_lrelu_4d(float* __restrict__ t, const float* __restrict__ meanv,
                              const float* __restrict__ invstd, int O) {
    size_t i = (size_t)blockIdx.x * blockDim.x + threadIdx.x;
    size_t tot = (size_t)B_ * O * N_ * KK_;
    if (i >= tot) return;
    int o = (int)((i / ((size_t)N_ * KK_)) % O);
    t[i] = lrelu((t[i] - meanv[o]) * invstd[o]);
}

// decode encoded-max [B,O,N], normalize, write into strided channel slab
__global__ void norm_max3d(const unsigned* __restrict__ emax, int O,
                           const float* __restrict__ meanv, const float* __restrict__ invstd,
                           float* __restrict__ dst, int c0out, int Cso) {
    int i = blockIdx.x * blockDim.x + threadIdx.x;
    if (i >= B_ * O * N_) return;
    int n = i % N_;
    int o = (i / N_) % O;
    int b = i / (N_ * O);
    float y = decf(emax[i]);
    dst[((size_t)b * Cso + c0out + o) * N_ + n] = lrelu((y - meanv[o]) * invstd[o]);
}

__global__ void norm_vec(const unsigned* __restrict__ emax, int O,
                         const float* __restrict__ meanv, const float* __restrict__ invstd,
                         float* __restrict__ dst) {
    int i = blockIdx.x * blockDim.x + threadIdx.x;
    if (i >= B_ * O) return;
    int o = i % O;
    dst[i] = lrelu((decf(emax[i]) - meanv[o]) * invstd[o]);
}

// --------------------------- transform-net tail ----------------------------

__global__ void fc_kernel(const float* __restrict__ in, const float* __restrict__ w,
                          const float* __restrict__ bias, float* __restrict__ out,
                          int O, int Cin) {
    int i = blockIdx.x * blockDim.x + threadIdx.x;
    if (i >= B_ * O) return;
    int b = i / O, o = i - b * O;
    const float* ir = in + (size_t)b * Cin;
    const float* wr = w  + (size_t)o * Cin;
    float s = bias ? bias[o] : 0.f;
    for (int c = 0; c < Cin; ++c) s += ir[c] * wr[c];
    out[i] = s;
}

__global__ void bn_batch_lrelu(float* __restrict__ g, int O) {   // BN over batch axis
    int o = blockIdx.x * blockDim.x + threadIdx.x;
    if (o >= O) return;
    float m = 0.f;
    for (int b = 0; b < B_; ++b) m += g[b * O + o];
    m *= (1.f / B_);
    float v = 0.f;
    for (int b = 0; b < B_; ++b) { float d = g[b * O + o] - m; v += d * d; }
    v *= (1.f / B_);
    float is = rsqrtf(v + EPSV);
    for (int b = 0; b < B_; ++b) g[b * O + o] = lrelu((g[b * O + o] - m) * is);
}

__global__ void fc3_makeT(const float* __restrict__ h, const float* __restrict__ w,
                          const float* __restrict__ bias, float* __restrict__ T) {
    int i = threadIdx.x;
    if (i >= B_ * 9) return;
    int b = i / 9, j = i - b * 9;
    float s = bias[j];
    for (int c = 0; c < 256; ++c) s += h[b * 256 + c] * w[j * 256 + c];
    if (j == 0 || j == 4 || j == 8) s += 1.f;      // + eye(3)
    T[i] = s;
}

__global__ void apply_T(const float* __restrict__ T, const float* __restrict__ x,
                        float* __restrict__ xo) {
    int i = blockIdx.x * blockDim.x + threadIdx.x;
    if (i >= B_ * 3 * N_) return;
    int n = i % N_;
    int r = (i / N_) % 3;
    int b = i / (3 * N_);
    float s = 0.f;
    for (int j = 0; j < 3; ++j)
        s += T[b * 9 + r * 3 + j] * x[((size_t)b * 3 + j) * N_ + n];
    xo[i] = s;
}

// ------------------------------ orchestration ------------------------------

extern "C" void kernel_launch(void* const* d_in, const int* in_sizes, int n_in,
                              void* d_out, int out_size, void* d_ws, size_t ws_size,
                              hipStream_t stream) {
    const float* x      = (const float*)d_in[0];
    const float* t_w1   = (const float*)d_in[1];
    const float* t_w2   = (const float*)d_in[2];
    const float* t_w3   = (const float*)d_in[3];
    const float* t_fc1  = (const float*)d_in[4];
    const float* t_fc2  = (const float*)d_in[5];
    const float* t_fc2b = (const float*)d_in[6];
    const float* t_fc3  = (const float*)d_in[7];
    const float* t_fc3b = (const float*)d_in[8];
    const float* w1     = (const float*)d_in[9];
    const float* w2     = (const float*)d_in[10];
    const float* w3     = (const float*)d_in[11];
    const float* w4     = (const float*)d_in[12];
    const float* w6     = (const float*)d_in[13];

    char* ws = (char*)d_ws;
    size_t off = 0;
    auto alloc = [&](size_t bytes) -> void* {
        void* p = ws + off;
        off = (off + bytes + 255) & ~(size_t)255;
        return p;
    };
    // bf16 weight copies
    unsigned short* wb_t1 = (unsigned short*)alloc(64 * 6 * 2);
    unsigned short* wb_t2 = (unsigned short*)alloc(128 * 64 * 2);
    unsigned short* wb_t3 = (unsigned short*)alloc(1024 * 128 * 2);
    unsigned short* wb1   = (unsigned short*)alloc(64 * 6 * 2);
    unsigned short* wb2   = (unsigned short*)alloc(64 * 128 * 2);
    unsigned short* wb3   = (unsigned short*)alloc(128 * 128 * 2);
    unsigned short* wb4   = (unsigned short*)alloc(256 * 256 * 2);
    unsigned short* wb6   = (unsigned short*)alloc(1024 * 512 * 2);
    // scratch tensors (~105 MB total)
    float*    xxv   = (float*)alloc((size_t)B_ * N_ * 4);
    float*    pd    = (float*)alloc((size_t)B_ * N_ * N_ * 4);
    int*      idxb  = (int*)alloc((size_t)B_ * N_ * KK_ * 4);
    float*    xT    = (float*)alloc((size_t)B_ * 3 * N_ * 4);
    float*    H1    = (float*)alloc((size_t)B_ * 64 * N_ * KK_ * 4);
    float*    Hm    = (float*)alloc((size_t)B_ * 128 * N_ * 4);
    float*    xcat  = (float*)alloc((size_t)B_ * 512 * N_ * 4);
    unsigned* emaxB = (unsigned*)alloc((size_t)B_ * 256 * N_ * 4);
    unsigned* emaxV = (unsigned*)alloc((size_t)B_ * 1024 * 4);
    float*    h0    = (float*)alloc((size_t)B_ * 1024 * 4);
    float*    h1v   = (float*)alloc((size_t)B_ * 512 * 4);
    float*    h2v   = (float*)alloc((size_t)B_ * 256 * 4);
    float*    T9    = (float*)alloc((size_t)B_ * 9 * 4);
    float*    ssum  = (float*)alloc(1024 * 4);
    float*    ssq   = (float*)alloc(1024 * 4);
    float*    smean = (float*)alloc(1024 * 4);
    float*    sinv  = (float*)alloc(1024 * 4);
    (void)ws_size; (void)in_sizes; (void)n_in; (void)out_size;

    auto cvt = [&](const float* src, unsigned short* dst, int n) {
        cvt_bf16<<<(n + 255) / 256, 256, 0, stream>>>(src, dst, n);
    };
    cvt(t_w1, wb_t1, 64 * 6);     cvt(t_w2, wb_t2, 128 * 64);
    cvt(t_w3, wb_t3, 1024 * 128); cvt(w1, wb1, 64 * 6);
    cvt(w2, wb2, 64 * 128);       cvt(w3, wb3, 128 * 128);
    cvt(w4, wb4, 256 * 256);      cvt(w6, wb6, 1024 * 512);

    auto gram_topk = [&](const float* base, int c0, int Cs, int C) {
        xx_kernel<<<(B_ * N_ + 255) / 256, 256, 0, stream>>>(base, c0, Cs, C, xxv);
        if ((C & 3) == 0)
            gram_wmma<true><<<dim3(N_ / 16, N_ / 16, B_), 32, 0, stream>>>(base, c0, Cs, C, xxv, pd);
        else
            gram_wmma<false><<<dim3(N_ / 16, N_ / 16, B_), 32, 0, stream>>>(base, c0, Cs, C, xxv, pd);
        topk20<<<B_ * N_, 128, 0, stream>>>(pd, idxb);
    };
    auto clear_stats = [&](int O) {
        hipMemsetAsync(ssum, 0, O * sizeof(float), stream);
        hipMemsetAsync(ssq,  0, O * sizeof(float), stream);
    };
    const int Pk = B_ * N_ * KK_;   // 163840
    const int Pn = B_ * N_;         // 8192

    // ======================= transform net =======================
    gram_topk(x, 0, 3, 3);
    // t-conv1: gather C=3 (Kc=6) -> full 4D activations
    clear_stats(64);
    edge_gemm<2, false><<<dim3(Pk / 64, 64 / 16), 128, 0, stream>>>(
        wb_t1, 64, 6, x, 0, 3, idxb, nullptr, H1, ssum, ssq);
    finalize_stats<<<1, 64, 0, stream>>>(ssum, ssq, (float)Pk, smean, sinv, 64);
    norm_lrelu_4d<<<(int)(((size_t)B_ * 64 * N_ * KK_ + 255) / 256), 256, 0, stream>>>(H1, smean, sinv, 64);
    // t-conv2: dense 4D (Kc=64), max over k  [async-LDS staged B]
    clear_stats(128);
    hipMemsetAsync(emaxB, 0, (size_t)B_ * 128 * N_ * 4, stream);
    dense_gemm<1, 0><<<dim3(Pk / 64, 128 / 16), 128, 0, stream>>>(
        wb_t2, 128, 64, H1, 0, 0, emaxB, ssum, ssq);
    finalize_stats<<<2, 64, 0, stream>>>(ssum, ssq, (float)Pk, smean, sinv, 128);
    norm_max3d<<<(B_ * 128 * N_ + 255) / 256, 256, 0, stream>>>(emaxB, 128, smean, sinv, Hm, 0, 128);
    // t_w3: dense 3D (Kc=128), max over n  [async-LDS staged B]
    clear_stats(1024);
    hipMemsetAsync(emaxV, 0, (size_t)B_ * 1024 * 4, stream);
    dense_gemm<2, 1><<<dim3(Pn / 64, 1024 / 16), 128, 0, stream>>>(
        wb_t3, 1024, 128, Hm, 0, 128, emaxV, ssum, ssq);
    finalize_stats<<<16, 64, 0, stream>>>(ssum, ssq, (float)Pn, smean, sinv, 1024);
    norm_vec<<<(B_ * 1024 + 255) / 256, 256, 0, stream>>>(emaxV, 1024, smean, sinv, h0);
    // fc chain -> T
    fc_kernel<<<(B_ * 512 + 255) / 256, 256, 0, stream>>>(h0, t_fc1, nullptr, h1v, 512, 1024);
    bn_batch_lrelu<<<2, 256, 0, stream>>>(h1v, 512);
    fc_kernel<<<(B_ * 256 + 255) / 256, 256, 0, stream>>>(h1v, t_fc2, t_fc2b, h2v, 256, 512);
    bn_batch_lrelu<<<1, 256, 0, stream>>>(h2v, 256);
    fc3_makeT<<<1, 128, 0, stream>>>(h2v, t_fc3, t_fc3b, T9);
    apply_T<<<(B_ * 3 * N_ + 255) / 256, 256, 0, stream>>>(T9, x, xT);

    // ======================= EdgeConv stack =======================
    auto edge_stage = [&](const unsigned short* wb, int O, const float* base,
                          int c0, int Cs, int Cin, int c0out) {
        gram_topk(base, c0, Cs, Cin);
        clear_stats(O);
        hipMemsetAsync(emaxB, 0, (size_t)B_ * O * N_ * 4, stream);
        if (((2 * Cin) & 31) == 0)
            edge_gemm<0, true><<<dim3(Pk / 64, O / 16), 128, 0, stream>>>(
                wb, O, 2 * Cin, base, c0, Cs, idxb, emaxB, nullptr, ssum, ssq);
        else
            edge_gemm<0, false><<<dim3(Pk / 64, O / 16), 128, 0, stream>>>(
                wb, O, 2 * Cin, base, c0, Cs, idxb, emaxB, nullptr, ssum, ssq);
        finalize_stats<<<(O + 63) / 64, 64, 0, stream>>>(ssum, ssq, (float)Pk, smean, sinv, O);
        norm_max3d<<<(B_ * O * N_ + 255) / 256, 256, 0, stream>>>(emaxB, O, smean, sinv, xcat, c0out, 512);
    };
    edge_stage(wb1,  64, xT,   0,   3,   3,   0);   // x1 -> xcat[:,  0: 64]
    edge_stage(wb2,  64, xcat, 0,   512, 64,  64);  // x2 -> xcat[:, 64:128]
    edge_stage(wb3, 128, xcat, 64,  512, 64,  128); // x3 -> xcat[:,128:256]
    edge_stage(wb4, 256, xcat, 128, 512, 128, 256); // x4 -> xcat[:,256:512]

    // final: w6 @ xcat (Kc=512), BN over (0,2), lrelu, max over N  [async-LDS]
    clear_stats(1024);
    hipMemsetAsync(emaxV, 0, (size_t)B_ * 1024 * 4, stream);
    dense_gemm<2, 1><<<dim3(Pn / 64, 1024 / 16), 128, 0, stream>>>(
        wb6, 1024, 512, xcat, 0, 512, emaxV, ssum, ssq);
    finalize_stats<<<16, 64, 0, stream>>>(ssum, ssq, (float)Pn, smean, sinv, 1024);
    norm_vec<<<(B_ * 1024 + 255) / 256, 256, 0, stream>>>(emaxV, 1024, smean, sinv, (float*)d_out);
}
